// TransformerBlock_10514079941222
// MI455X (gfx1250) — compile-verified, matrix operand order checked by reading, not run on hardware
//
#include <hip/hip_runtime.h>
#include <hip/hip_bf16.h>
#include <math.h>

typedef __attribute__((ext_vector_type(16))) _Float16 v16h;
typedef __attribute__((ext_vector_type(8)))  _Float16 h8;
typedef __attribute__((ext_vector_type(8)))  float    v8f;

#define D_MODEL 1024
#define N_HEADS 16
#define DKH     64
#define D_FF    4096
#define N_EXPERTS 8
#define BATCH   2
#define SEQ     2048
#define TOKENS  (BATCH*SEQ)
#define LN_EPS  1e-5f

// Use CDNA5 async global->LDS copies (ASYNCcnt) for A-tile staging in the GEMM.
#define USE_ASYNC_LDS 1

// ---------------------------------------------------------------- helpers
__device__ __forceinline__ v8f wmma_f16(v16h a, v16h b, v8f c) {
  return __builtin_amdgcn_wmma_f32_16x16x32_f16(false, a, false, b, (short)0, c, false, false);
}

// Load a 16-wide f16 fragment: lanes 0-15 take p[8*hi..+7] then p[16+8*hi..+7]
// (hi = lane>>4), matching the CDNA5 16-bit A/B VGPR layout over 32 contiguous K.
__device__ __forceinline__ v16h ld_frag(const _Float16* p, int hi) {
  union { v16h v; h8 h[2]; } u;
  u.h[0] = *(const h8*)(p + 8 * hi);
  u.h[1] = *(const h8*)(p + 16 + 8 * hi);
  return u.v;
}

__device__ __forceinline__ float gelu_f(float x) {
  return 0.5f * x * (1.0f + erff(x * 0.70710678118654752f));
}
__device__ __forceinline__ float silu_f(float x) {
  return x / (1.0f + __expf(-x));
}

#if USE_ASYNC_LDS
// Async copy of 16 bytes global->LDS per lane; tracked by ASYNCcnt.
__device__ __forceinline__ void async_copy_b128(unsigned int lds_addr, const _Float16* gsrc) {
  asm volatile("global_load_async_to_lds_b128 %0, %1, off"
               :: "v"(lds_addr), "v"(gsrc) : "memory");
}
#if defined(__has_builtin)
#if __has_builtin(__builtin_amdgcn_s_wait_asynccnt)
#define WAIT_ASYNC() __builtin_amdgcn_s_wait_asynccnt(0)
#endif
#endif
#ifndef WAIT_ASYNC
#define WAIT_ASYNC() asm volatile("s_wait_asynccnt 0" ::: "memory")
#endif
__device__ __forceinline__ unsigned int lds_addr_of(const void* p) {
  return (unsigned int)(unsigned long long)p;   // generic LDS ptr: addr[31:0] = LDS offset
}
#endif

// ---------------------------------------------------------------- f32 -> f16 convert
__global__ void cvt16_kernel(const float* __restrict__ src, _Float16* __restrict__ dst, size_t n) {
  size_t i = (size_t)blockIdx.x * blockDim.x + threadIdx.x;
  size_t stride = (size_t)gridDim.x * blockDim.x;
  for (; i < n; i += stride) dst[i] = (_Float16)src[i];
}

// ---------------------------------------------------------------- generic WMMA GEMM
// C[M,N](f32 or f16) = act(A[M,K]*B[K,N] + bias) * scale
// BM=128, BN=64, BK=32; 256 threads = 8 waves (4x2); each wave: 32 rows x 32 cols
// = 4 WMMAs / K-step. A-tiles double-buffered in LDS and filled with async
// global->LDS copies one K-step ahead; B-tiles register-pipelined + transposed.
// seg_off/seg_idx: dynamic M segment (MoE); gather: A-row indirection.
__global__ __launch_bounds__(256)
void gemm_wmma_kernel(const _Float16* __restrict__ A, int lda,
                      const _Float16* __restrict__ Bm, int ldb,
                      const float* __restrict__ bias,
                      float* __restrict__ Cf, _Float16* __restrict__ Ch, int ldc,
                      int M, int K,
                      const int* __restrict__ seg_off, int seg_idx,
                      const int* __restrict__ gather,
                      int act, float scale) {
  int base = 0, Mloc = M;
  if (seg_off) { base = seg_off[seg_idx]; Mloc = seg_off[seg_idx + 1] - base; }
  const int tm = blockIdx.y, tn = blockIdx.x;
  if (tm * 128 >= Mloc) return;

  __shared__ __align__(16) _Float16 As[2][128 * 32];  // [buf][m][k]
  __shared__ __align__(16) _Float16 Bs[64 * 32];      // transposed: [n][k]
  __shared__ int rowA[128];

  const int tid = threadIdx.x;
  if (tid < 128) {
    int m = tm * 128 + tid;
    if (m >= Mloc) m = Mloc - 1;          // duplicate last valid row for edge tiles
    int g = base + m;
    rowA[tid] = gather ? gather[g] : (seg_off ? g : m);
  }
  __syncthreads();

  const int lane = tid & 31, wv = tid >> 5;
  const int wr = wv >> 1, wc = wv & 1;           // wave tile: rows 32*wr, cols 32*wc
  const int l15 = lane & 15, hi = lane >> 4;

  // A loader: each thread owns row tid>>1, 16 contiguous halves at (tid&1)*16.
  const int aRowL = tid >> 1, aKc = (tid & 1) * 16;
  const int myArow = rowA[aRowL];
  const _Float16* aG = A + (size_t)myArow * lda + aKc;
  // B loader: each thread owns k-row tid>>3, 8 halves at (tid&7)*8.
  const int bkk = tid >> 3, bnn = (tid & 7) * 8;
  const _Float16* bG = Bm + (size_t)bkk * ldb + (size_t)tn * 64 + bnn;

#if USE_ASYNC_LDS
  const unsigned int aL0 = lds_addr_of(&As[0][aRowL * 32 + aKc]);
  const unsigned int aL1 = lds_addr_of(&As[1][aRowL * 32 + aKc]);
#endif

  v8f c[2][2] = {};

  // ---- prologue: stage k0 = 0
#if USE_ASYNC_LDS
  async_copy_b128(aL0,     aG);
  async_copy_b128(aL0 + 16, aG + 8);
#else
  h8 avc0 = *(const h8*)(aG);
  h8 avc1 = *(const h8*)(aG + 8);
#endif
  h8 bv0 = *(const h8*)(bG);

  const int nsteps = K / 32;
  for (int i = 0; i < nsteps; ++i) {
    const int cur = i & 1;
    // commit B registers into LDS (transposed [n][k])
#pragma unroll
    for (int j = 0; j < 8; ++j) Bs[(bnn + j) * 32 + bkk] = bv0[j];
#if USE_ASYNC_LDS
    WAIT_ASYNC();                       // A tile (buf cur) resident
#else
    *(h8*)(&As[cur][aRowL * 32 + aKc])     = avc0;
    *(h8*)(&As[cur][aRowL * 32 + aKc + 8]) = avc1;
#endif
    __syncthreads();

    // kick off next K-step's loads while WMMAs run on the current tile
    if (i + 1 < nsteps) {
      const int kn = (i + 1) * 32;
#if USE_ASYNC_LDS
      const unsigned int aLn = (cur ? aL0 : aL1);
      async_copy_b128(aLn,      aG + kn);
      async_copy_b128(aLn + 16, aG + kn + 8);
#else
      avc0 = *(const h8*)(aG + kn);
      avc1 = *(const h8*)(aG + kn + 8);
#endif
      bv0 = *(const h8*)(bG + (size_t)kn * ldb);
    }

    v16h a0 = ld_frag(&As[cur][(wr * 32 + l15) * 32], hi);
    v16h a1 = ld_frag(&As[cur][(wr * 32 + 16 + l15) * 32], hi);
    v16h b0 = ld_frag(&Bs[(wc * 32 + l15) * 32], hi);
    v16h b1 = ld_frag(&Bs[(wc * 32 + 16 + l15) * 32], hi);
    c[0][0] = wmma_f16(a0, b0, c[0][0]);
    c[0][1] = wmma_f16(a0, b1, c[0][1]);
    c[1][0] = wmma_f16(a1, b0, c[1][0]);
    c[1][1] = wmma_f16(a1, b1, c[1][1]);
    __syncthreads();
  }

  const int colBase = tn * 64 + wc * 32 + l15;
#pragma unroll
  for (int mi = 0; mi < 2; ++mi) {
    const int cmBase = tm * 128 + wr * 32 + mi * 16 + 8 * hi;
#pragma unroll
    for (int r = 0; r < 8; ++r) {
      int ml = cmBase + r;
      if (ml >= Mloc) continue;
      size_t crow = (size_t)(base + ml);
#pragma unroll
      for (int ni = 0; ni < 2; ++ni) {
        int col = colBase + ni * 16;
        float v = c[mi][ni][r] + (bias ? bias[col] : 0.0f);
        if (act == 1)      v = gelu_f(v);
        else if (act == 2) v = silu_f(v);
        v *= scale;
        if (Ch) Ch[crow * ldc + col] = (_Float16)v;
        else    Cf[crow * ldc + col] = v;
      }
    }
  }
}

// ---------------------------------------------------------------- flash attention
// 1 wave per 16-query tile; 2 waves per block. Q pre-scaled by 1/sqrt(dk).
// Streams keys in 32-blocks: 4 score WMMAs + online softmax + 4 ctx WMMAs.
__global__ __launch_bounds__(64)
void flash_attn_kernel(const _Float16* __restrict__ qh,
                       const _Float16* __restrict__ kh,
                       const _Float16* __restrict__ vh,
                       _Float16* __restrict__ oh) {
  __shared__ __align__(16) _Float16 Plds[2][16 * 32];  // [query][key32] per wave
  __shared__ __align__(16) _Float16 Vt[2][64 * 32];    // [dk][key32]   per wave

  const int tid = threadIdx.x;
  const int wv = tid >> 5, lane = tid & 31;
  const int l15 = lane & 15, hi = lane >> 4;
  const int qt = blockIdx.x * 2 + wv;
  const int b  = qt >> 11;            // 2048 q-tiles per batch (H*S/16)
  const int hH = (qt >> 7) & 15;
  const int q0 = (qt & 127) << 4;
  const size_t D = D_MODEL;
  const size_t hoff = (size_t)hH * DKH;

  v16h qa[2];
  {
    size_t qrow = ((size_t)(b * SEQ + q0 + l15)) * D + hoff;
#pragma unroll
    for (int ks = 0; ks < 2; ++ks) {
      union { v16h v; h8 h[2]; } u;
      u.h[0] = *(const h8*)(qh + qrow + ks * 32 + 8 * hi);
      u.h[1] = *(const h8*)(qh + qrow + ks * 32 + 16 + 8 * hi);
      qa[ks] = u.v;
    }
  }

  float om[8], ls[8];
#pragma unroll
  for (int r = 0; r < 8; ++r) { om[r] = -3.0e38f; ls[r] = 0.0f; }
  v8f ctx[4] = {};

  for (int kb = 0; kb < SEQ / 32; ++kb) {
    const int kbase = kb * 32;
    v8f s[2];
#pragma unroll
    for (int g = 0; g < 2; ++g) {
      size_t krow = ((size_t)(b * SEQ + kbase + g * 16 + l15)) * D + hoff;
      v8f acc = {};
#pragma unroll
      for (int ks = 0; ks < 2; ++ks) {
        union { v16h v; h8 h[2]; } u;
        u.h[0] = *(const h8*)(kh + krow + ks * 32 + 8 * hi);
        u.h[1] = *(const h8*)(kh + krow + ks * 32 + 16 + 8 * hi);
        acc = wmma_f16(qa[ks], u.v, acc);
      }
      s[g] = acc;
    }
    {
      size_t vrow = ((size_t)(b * SEQ + kbase + lane)) * D + hoff;
#pragma unroll
      for (int cch = 0; cch < 8; ++cch) {
        h8 vvec = *(const h8*)(vh + vrow + cch * 8);
#pragma unroll
        for (int j = 0; j < 8; ++j) Vt[wv][(cch * 8 + j) * 32 + lane] = vvec[j];
      }
    }
#pragma unroll
    for (int r = 0; r < 8; ++r) {
      float mr = fmaxf(s[0][r], s[1][r]);
#pragma unroll
      for (int m = 1; m < 16; m <<= 1) mr = fmaxf(mr, __shfl_xor(mr, m, 32));
      float nm = fmaxf(om[r], mr);
      float al = __expf(om[r] - nm);
      om[r] = nm;
      float p0 = __expf(s[0][r] - nm);
      float p1 = __expf(s[1][r] - nm);
      float rs = p0 + p1;
#pragma unroll
      for (int m = 1; m < 16; m <<= 1) rs += __shfl_xor(rs, m, 32);
      ls[r] = ls[r] * al + rs;
#pragma unroll
      for (int t = 0; t < 4; ++t) ctx[t][r] *= al;
      Plds[wv][(r + 8 * hi) * 32 + l15]      = (_Float16)p0;
      Plds[wv][(r + 8 * hi) * 32 + 16 + l15] = (_Float16)p1;
    }
    __syncthreads();
    union { v16h v; h8 h[2]; } up;
    up.h[0] = *(const h8*)(&Plds[wv][l15 * 32 + 8 * hi]);
    up.h[1] = *(const h8*)(&Plds[wv][l15 * 32 + 16 + 8 * hi]);
#pragma unroll
    for (int t = 0; t < 4; ++t) {
      union { v16h v; h8 h[2]; } ub;
      const _Float16* vp = &Vt[wv][(t * 16 + l15) * 32];
      ub.h[0] = *(const h8*)(vp + 8 * hi);
      ub.h[1] = *(const h8*)(vp + 16 + 8 * hi);
      ctx[t] = wmma_f16(up.v, ub.v, ctx[t]);
    }
    __syncthreads();
  }
#pragma unroll
  for (int r = 0; r < 8; ++r) {
    float inv = 1.0f / ls[r];
    size_t orow = ((size_t)(b * SEQ + q0 + 8 * hi + r)) * D + hoff;
#pragma unroll
    for (int t = 0; t < 4; ++t)
      oh[orow + t * 16 + l15] = (_Float16)(ctx[t][r] * inv);
  }
}

// ---------------------------------------------------------------- residual + LayerNorm
__global__ __launch_bounds__(256)
void ln_fused_kernel(const float* __restrict__ xa, const float* __restrict__ xb,
                     const float* __restrict__ g, const float* __restrict__ bb,
                     float* __restrict__ of, _Float16* __restrict__ oh) {
  __shared__ float red[256];
  const int t = blockIdx.x, tid = threadIdx.x;
  const int D = D_MODEL;
  float h[4];
  float s = 0.0f;
#pragma unroll
  for (int i = 0; i < 4; ++i) {
    int d = tid + i * 256;
    float v = xa[(size_t)t * D + d];
    if (xb) v += xb[(size_t)t * D + d];
    h[i] = v; s += v;
  }
  red[tid] = s; __syncthreads();
  for (int o = 128; o; o >>= 1) { if (tid < o) red[tid] += red[tid + o]; __syncthreads(); }
  float mean = red[0] * (1.0f / D);
  __syncthreads();
  float vs = 0.0f;
#pragma unroll
  for (int i = 0; i < 4; ++i) { float dd = h[i] - mean; vs += dd * dd; }
  red[tid] = vs; __syncthreads();
  for (int o = 128; o; o >>= 1) { if (tid < o) red[tid] += red[tid + o]; __syncthreads(); }
  float rstd = rsqrtf(red[0] * (1.0f / D) + LN_EPS);
#pragma unroll
  for (int i = 0; i < 4; ++i) {
    int d = tid + i * 256;
    float o = (h[i] - mean) * rstd * g[d] + bb[d];
    of[(size_t)t * D + d] = o;
    if (oh) oh[(size_t)t * D + d] = (_Float16)o;
  }
}

// ---------------------------------------------------------------- gating: logits + top-2 softmax
__global__ __launch_bounds__(128)
void gate_topk_kernel(const float* __restrict__ x1, const float* __restrict__ gw,
                      const float* __restrict__ gb,
                      int* __restrict__ tok_e, float* __restrict__ tok_w) {
  const int wv = threadIdx.x >> 5, lane = threadIdx.x & 31;
  const int t = blockIdx.x * 4 + wv;
  float acc[N_EXPERTS];
#pragma unroll
  for (int e = 0; e < N_EXPERTS; ++e) acc[e] = 0.0f;
  for (int d = lane; d < D_MODEL; d += 32) {
    float xv = x1[(size_t)t * D_MODEL + d];
#pragma unroll
    for (int e = 0; e < N_EXPERTS; ++e) acc[e] += xv * gw[d * N_EXPERTS + e];
  }
#pragma unroll
  for (int e = 0; e < N_EXPERTS; ++e)
#pragma unroll
    for (int m = 1; m < 32; m <<= 1) acc[e] += __shfl_xor(acc[e], m, 32);
  if (lane == 0) {
    float lg[N_EXPERTS];
#pragma unroll
    for (int e = 0; e < N_EXPERTS; ++e) lg[e] = acc[e] + gb[e];
    int e0 = 0;
    for (int e = 1; e < N_EXPERTS; ++e) if (lg[e] > lg[e0]) e0 = e;   // ties -> lower idx
    int e1 = -1;
    for (int e = 0; e < N_EXPERTS; ++e)
      if (e != e0 && (e1 < 0 || lg[e] > lg[e1])) e1 = e;
    float mx = fmaxf(lg[e0], lg[e1]);
    float w0 = __expf(lg[e0] - mx), w1 = __expf(lg[e1] - mx);
    float inv = 1.0f / (w0 + w1);
    tok_e[t * 2] = e0;  tok_e[t * 2 + 1] = e1;
    tok_w[t * 2] = w0 * inv;  tok_w[t * 2 + 1] = w1 * inv;
  }
}

// ---------------------------------------------------------------- routing
__global__ void zero8_kernel(int* counts, int* cursor) {
  if (threadIdx.x < N_EXPERTS) { counts[threadIdx.x] = 0; cursor[threadIdx.x] = 0; }
}
__global__ void count_e_kernel(const int* __restrict__ tok_e, int* counts, int n) {
  int i = blockIdx.x * blockDim.x + threadIdx.x;
  if (i < n) atomicAdd(&counts[tok_e[i]], 1);
}
__global__ void prefix8_kernel(const int* __restrict__ counts, int* seg) {
  if (threadIdx.x == 0) {
    int s = 0;
    for (int e = 0; e < N_EXPERTS; ++e) { seg[e] = s; s += counts[e]; }
    seg[N_EXPERTS] = s;
  }
}
__global__ void scatter_e_kernel(const int* __restrict__ tok_e, const int* __restrict__ seg,
                                 int* cursor, int* __restrict__ slot_token,
                                 int* __restrict__ tok_slot, int n) {
  int i = blockIdx.x * blockDim.x + threadIdx.x;
  if (i >= n) return;
  int e = tok_e[i];
  int pos = atomicAdd(&cursor[e], 1);
  int slot = seg[e] + pos;
  slot_token[slot] = i >> 1;
  tok_slot[i] = slot;
}

// ---------------------------------------------------------------- weighted combine of expert outputs
__global__ __launch_bounds__(256)
void moe_combine_kernel(const float* __restrict__ y, const int* __restrict__ tok_slot,
                        const float* __restrict__ tok_w, float* __restrict__ out) {
  const int t = blockIdx.x, tid = threadIdx.x;
  const size_t s0 = (size_t)tok_slot[t * 2], s1 = (size_t)tok_slot[t * 2 + 1];
  const float w0 = tok_w[t * 2], w1 = tok_w[t * 2 + 1];
#pragma unroll
  for (int i = 0; i < 4; ++i) {
    int d = tid + i * 256;
    out[(size_t)t * D_MODEL + d] =
        w0 * y[s0 * D_MODEL + d] + w1 * y[s1 * D_MODEL + d];
  }
}

// ---------------------------------------------------------------- host
extern "C" void kernel_launch(void* const* d_in, const int* in_sizes, int n_in,
                              void* d_out, int out_size, void* d_ws, size_t ws_size,
                              hipStream_t stream) {
  (void)in_sizes; (void)n_in; (void)out_size; (void)ws_size;
  const float* x    = (const float*)d_in[0];
  const float* Wq   = (const float*)d_in[1];
  const float* bq   = (const float*)d_in[2];
  const float* Wk   = (const float*)d_in[3];
  const float* bk   = (const float*)d_in[4];
  const float* Wv   = (const float*)d_in[5];
  const float* bv   = (const float*)d_in[6];
  const float* Wo   = (const float*)d_in[7];
  const float* bo   = (const float*)d_in[8];
  const float* ln1g = (const float*)d_in[9];
  const float* ln1b = (const float*)d_in[10];
  const float* gw   = (const float*)d_in[11];
  const float* gb   = (const float*)d_in[12];
  const float* W1   = (const float*)d_in[13];
  const float* b1   = (const float*)d_in[14];
  const float* W2   = (const float*)d_in[15];
  const float* b2   = (const float*)d_in[16];
  const float* ln3g = (const float*)d_in[17];
  const float* ln3b = (const float*)d_in[18];

  const size_t T = TOKENS, D = D_MODEL, F = D_FF, E = N_EXPERTS;
  char* ws = (char*)d_ws;
  size_t off = 0;
  auto alloc = [&](size_t bytes) -> char* {
    char* p = ws + off;
    off = (off + bytes + 255) & ~(size_t)255;
    return p;
  };
  _Float16* xh   = (_Float16*)alloc(T * D * 2);
  _Float16* wqh  = (_Float16*)alloc(D * D * 2);
  _Float16* wkh  = (_Float16*)alloc(D * D * 2);
  _Float16* wvh  = (_Float16*)alloc(D * D * 2);
  _Float16* woh  = (_Float16*)alloc(D * D * 2);
  _Float16* w1h  = (_Float16*)alloc(E * D * F * 2);
  _Float16* w2h  = (_Float16*)alloc(E * F * D * 2);
  _Float16* qh   = (_Float16*)alloc(T * D * 2);
  _Float16* kh   = (_Float16*)alloc(T * D * 2);
  _Float16* vh   = (_Float16*)alloc(T * D * 2);
  _Float16* ctxh = (_Float16*)alloc(T * D * 2);
  float*    attn = (float*)   alloc(T * D * 4);
  float*    x1   = (float*)   alloc(T * D * 4);
  _Float16* x1h  = (_Float16*)alloc(T * D * 2);
  int*      tok_e= (int*)     alloc(2 * T * 4);
  float*    tok_w= (float*)   alloc(2 * T * 4);
  int*      counts=(int*)     alloc(16 * 4);
  int*      cursor=(int*)     alloc(16 * 4);
  int*      seg  = (int*)     alloc(16 * 4);
  int*      slot_token=(int*) alloc(2 * T * 4);
  int*      tok_slot  =(int*) alloc(2 * T * 4);
  _Float16* h_act= (_Float16*)alloc(2 * T * F * 2);
  float*    yb   = (float*)   alloc(2 * T * D * 4);
  float*    moe  = (float*)   alloc(T * D * 4);

  // f32 -> f16 conversions (activations + weights)
  cvt16_kernel<<<4096, 256, 0, stream>>>(x,  xh,  T * D);
  cvt16_kernel<<<2048, 256, 0, stream>>>(Wq, wqh, D * D);
  cvt16_kernel<<<2048, 256, 0, stream>>>(Wk, wkh, D * D);
  cvt16_kernel<<<2048, 256, 0, stream>>>(Wv, wvh, D * D);
  cvt16_kernel<<<2048, 256, 0, stream>>>(Wo, woh, D * D);
  cvt16_kernel<<<8192, 256, 0, stream>>>(W1, w1h, E * D * F);
  cvt16_kernel<<<8192, 256, 0, stream>>>(W2, w2h, E * F * D);

  // QKV projections (Q pre-scaled by 1/sqrt(dk) = 0.125)
  dim3 gQKV(D / 64, T / 128);
  gemm_wmma_kernel<<<gQKV, 256, 0, stream>>>(xh, (int)D, wqh, (int)D, bq,
      nullptr, qh, (int)D, (int)T, (int)D, nullptr, 0, nullptr, 0, 0.125f);
  gemm_wmma_kernel<<<gQKV, 256, 0, stream>>>(xh, (int)D, wkh, (int)D, bk,
      nullptr, kh, (int)D, (int)T, (int)D, nullptr, 0, nullptr, 0, 1.0f);
  gemm_wmma_kernel<<<gQKV, 256, 0, stream>>>(xh, (int)D, wvh, (int)D, bv,
      nullptr, vh, (int)D, (int)T, (int)D, nullptr, 0, nullptr, 0, 1.0f);

  // flash attention: B*H*(S/16) = 4096 q-tiles, 2 per block
  flash_attn_kernel<<<2048, 64, 0, stream>>>(qh, kh, vh, ctxh);

  // output projection
  gemm_wmma_kernel<<<gQKV, 256, 0, stream>>>(ctxh, (int)D, woh, (int)D, bo,
      attn, nullptr, (int)D, (int)T, (int)D, nullptr, 0, nullptr, 0, 1.0f);

  // x1 = LN(x + attn), also f16 copy
  ln_fused_kernel<<<(int)T, 256, 0, stream>>>(x, attn, ln1g, ln1b, x1, x1h);

  // gating + top-2 routing
  gate_topk_kernel<<<(int)(T / 4), 128, 0, stream>>>(x1, gw, gb, tok_e, tok_w);
  zero8_kernel<<<1, 32, 0, stream>>>(counts, cursor);
  count_e_kernel<<<(int)((2 * T + 255) / 256), 256, 0, stream>>>(tok_e, counts, (int)(2 * T));
  prefix8_kernel<<<1, 32, 0, stream>>>(counts, seg);
  scatter_e_kernel<<<(int)((2 * T + 255) / 256), 256, 0, stream>>>(
      tok_e, seg, cursor, slot_token, tok_slot, (int)(2 * T));

  // sparse MoE: per-expert GEMMs over dynamic token segments
  for (int e = 0; e < (int)E; ++e) {
    gemm_wmma_kernel<<<dim3(F / 64, T / 128), 256, 0, stream>>>(
        x1h, (int)D, w1h + (size_t)e * D * F, (int)F, b1 + (size_t)e * F,
        nullptr, h_act, (int)F, 0, (int)D, seg, e, slot_token,
        (e % 2 == 0) ? 1 : 2, 1.0f);
    gemm_wmma_kernel<<<dim3(D / 64, T / 128), 256, 0, stream>>>(
        h_act, (int)F, w2h + (size_t)e * F * D, (int)D, b2 + (size_t)e * D,
        yb, nullptr, (int)D, 0, (int)F, seg, e, nullptr, 0, 1.0f);
  }

  // combine top-2 expert outputs per token
  moe_combine_kernel<<<(int)T, 256, 0, stream>>>(yb, tok_slot, tok_w, moe);

  // out = LN(x1 + moe)
  ln_fused_kernel<<<(int)T, 256, 0, stream>>>(x1, moe, ln3g, ln3b, (float*)d_out, nullptr);
}
